// DGNRNetwork_29798483100455
// MI455X (gfx1250) — matrix operand order repaired and checked by reading, not compiled, not run on hardware
//
#include <hip/hip_runtime.h>
#include <hip/hip_bf16.h>

// Problem constants (match reference setup_inputs)
#define NN     50000
#define EE     800000
#define F_INN  128
#define HHH    32
#define HDD    128     // HEADS*H
#define BBB    64
#define QHID   128

typedef __attribute__((ext_vector_type(16))) __bf16 v16bf;
typedef __attribute__((ext_vector_type(8)))  float  v8f;

// ---------------------------------------------------------------------------
// Generic GEMM: C[rows x Fout] = relu?(A[rows x K] @ W[K x Fout] + bias)
// One wave (32 lanes) per 16x16 output tile, K stepped by 32 using
// v_wmma_f32_16x16x32_bf16 (bf16 inputs, f32 accumulate).
// A fragment layout (16-bit A, 16x32): lane<16 -> M=lane, K = i(+8 high half);
// lanes 16..31 -> same M rows, K offset by 8 within each 16-wide group.
// B fragment (16-bit, 32x16): lanes 0-15 N=lane K=0..7,16..23, lanes 16-31 K=8..15,24..31 (same as A).
// C/D: VGPR r -> M = r + (lane>=16 ? 8 : 0), N = lane&15.
// ---------------------------------------------------------------------------
__global__ void gemm_bf16_wmma(const float* __restrict__ A,
                               const float* __restrict__ W,
                               const float* __restrict__ bias,
                               float* __restrict__ C,
                               int K, int Fout, int doRelu)
{
    const int tileM = blockIdx.x;
    const int tileN = blockIdx.y;
    const int lane  = threadIdx.x;        // 0..31, all lanes active (EXEC all 1s)
    const int half  = lane >> 4;
    const int l15   = lane & 15;
    const int rowA  = tileM * 16 + l15;
    const int colB  = tileN * 16 + l15;

    v8f acc = {};
    for (int kk = 0; kk < K; kk += 32) {
        v16bf af, bf_;
        const float* arow = A + (size_t)rowA * K + kk + half * 8;
#pragma unroll
        for (int i = 0; i < 8; ++i) af[i]     = (__bf16)arow[i];
#pragma unroll
        for (int i = 0; i < 8; ++i) af[8 + i] = (__bf16)arow[16 + i];

        const float* wcol = W + (size_t)(kk + half * 8) * Fout + colB;
#pragma unroll
        for (int i = 0; i < 16; ++i) bf_[i] = (__bf16)wcol[(size_t)(i < 8 ? i : i + 8) * Fout];

        acc = __builtin_amdgcn_wmma_f32_16x16x32_bf16(
                  false, af, false, bf_, (short)0, acc, false, false);
    }

    const float bv = bias[colB];
#pragma unroll
    for (int r = 0; r < 8; ++r) {
        int m = tileM * 16 + r + half * 8;
        float o = acc[r] + bv;
        if (doRelu) o = fmaxf(o, 0.0f);
        C[(size_t)m * Fout + colB] = o;
    }
}

// ---------------------------------------------------------------------------
// Edge attention passes
// ---------------------------------------------------------------------------
__device__ __forceinline__ void atomicMaxF(float* addr, float val)
{
    if (val >= 0.0f)
        atomicMax((int*)addr, __float_as_int(val));
    else
        atomicMin((unsigned int*)addr, __float_as_uint(val));
}

__global__ void init_neg_inf_kernel(float* __restrict__ p, int n)
{
    int t = blockIdx.x * blockDim.x + threadIdx.x;
    if (t < n) p[t] = -INFINITY;
}

// Pass A: one wave per edge. Lane l loads 4 channels (float4) of q[dst], k[src];
// 8-lane shuffle reduction -> per-head logit; segment max via atomics.
__global__ void edge_logits_kernel(const float* __restrict__ q,
                                   const float* __restrict__ k,
                                   const int* __restrict__ ei,
                                   float* __restrict__ logits,
                                   float* __restrict__ m)
{
    int t = blockIdx.x * blockDim.x + threadIdx.x;
    int e = t >> 5;
    if (e >= EE) return;
    int lane = t & 31;
    int src = ei[e];
    int dst = ei[EE + e];

    const float4 qv = *(const float4*)(q + (size_t)dst * HDD + lane * 4);
    const float4 kv = *(const float4*)(k + (size_t)src * HDD + lane * 4);
    float s = qv.x * kv.x + qv.y * kv.y + qv.z * kv.z + qv.w * kv.w;
    s += __shfl_xor(s, 1, 32);
    s += __shfl_xor(s, 2, 32);
    s += __shfl_xor(s, 4, 32);   // sum over 8 lanes = 32 channels of one head
    if ((lane & 7) == 0) {
        int h = lane >> 3;
        s *= 0.17677669529663687f;          // 1/sqrt(32)
        logits[(size_t)e * 4 + h] = s;
        atomicMaxF(&m[(size_t)dst * 4 + h], s);
    }
}

// Pass B: a = exp(logit - max); denom[dst,h] += a; accum[dst,:] += a * v[src,:]
__global__ void edge_accum_kernel(const float* __restrict__ v,
                                  const int* __restrict__ ei,
                                  const float* __restrict__ logits,
                                  const float* __restrict__ m,
                                  float* __restrict__ denom,
                                  float* __restrict__ accum)
{
    int t = blockIdx.x * blockDim.x + threadIdx.x;
    int e = t >> 5;
    if (e >= EE) return;
    int lane = t & 31;
    int h = lane >> 3;
    int src = ei[e];
    int dst = ei[EE + e];

    float a = __expf(logits[(size_t)e * 4 + h] - m[(size_t)dst * 4 + h]);
    if ((lane & 7) == 0)
        atomicAdd(&denom[(size_t)dst * 4 + h], a);

    const float4 vv = *(const float4*)(v + (size_t)src * HDD + lane * 4);
    float* ap = accum + (size_t)dst * HDD + lane * 4;
    atomicAdd(ap + 0, a * vv.x);
    atomicAdd(ap + 1, a * vv.y);
    atomicAdd(ap + 2, a * vv.z);
    atomicAdd(ap + 3, a * vv.w);
}

// out = relu(accum / (denom + eps))
__global__ void finalize_kernel(const float* __restrict__ accum,
                                const float* __restrict__ denom,
                                float* __restrict__ out)
{
    int t = blockIdx.x * blockDim.x + threadIdx.x;
    if (t >= NN * HDD) return;
    int n = t >> 7;
    int c = t & 127;
    float d = denom[(size_t)n * 4 + (c >> 5)] + 1e-16f;
    out[t] = fmaxf(accum[t] / d, 0.0f);
}

// ---------------------------------------------------------------------------
// Final Q-head MLP: z=[x1(32)|x2(128)|x3(128)] -> relu(z@W1+b1) @ W2 + b2
// One block (128 threads) per graph.
// ---------------------------------------------------------------------------
__global__ void qmlp_kernel(const float* __restrict__ hEnc,
                            const float* __restrict__ h2,
                            const float* __restrict__ h3,
                            const int* __restrict__ idx,
                            const float* __restrict__ w1, const float* __restrict__ b1,
                            const float* __restrict__ w2, const float* __restrict__ b2,
                            float* __restrict__ out)
{
    __shared__ float z[288];
    __shared__ float hid[QHID];
    int b = blockIdx.x;
    int t = threadIdx.x;          // 0..127
    int node = idx[b];
    if (t < 32) z[t] = hEnc[(size_t)node * HHH + t];
    z[32 + t]  = h2[(size_t)node * HDD + t];
    z[160 + t] = h3[(size_t)node * HDD + t];
    __syncthreads();

    float s = b1[t];
    for (int kx = 0; kx < 288; ++kx)
        s = fmaf(z[kx], w1[(size_t)kx * QHID + t], s);
    hid[t] = fmaxf(s, 0.0f);
    __syncthreads();

    if (t < 2) {
        float o = b2[t];
        for (int j = 0; j < QHID; ++j)
            o = fmaf(hid[j], w2[j * 2 + t], o);
        out[b * 2 + t] = o;
    }
}

// ---------------------------------------------------------------------------
extern "C" void kernel_launch(void* const* d_in, const int* in_sizes, int n_in,
                              void* d_out, int out_size, void* d_ws, size_t ws_size,
                              hipStream_t stream)
{
    const float* x      = (const float*)d_in[0];
    const int*   ei     = (const int*)  d_in[1];
    const int*   idx    = (const int*)  d_in[2];
    const float* enc_w1 = (const float*)d_in[3];
    const float* enc_b1 = (const float*)d_in[4];
    const float* enc_w2 = (const float*)d_in[5];
    const float* enc_b2 = (const float*)d_in[6];
    const float* c1_wq = (const float*)d_in[7];  const float* c1_bq = (const float*)d_in[8];
    const float* c1_wk = (const float*)d_in[9];  const float* c1_bk = (const float*)d_in[10];
    const float* c1_wv = (const float*)d_in[11]; const float* c1_bv = (const float*)d_in[12];
    const float* c2_wq = (const float*)d_in[13]; const float* c2_bq = (const float*)d_in[14];
    const float* c2_wk = (const float*)d_in[15]; const float* c2_bk = (const float*)d_in[16];
    const float* c2_wv = (const float*)d_in[17]; const float* c2_bv = (const float*)d_in[18];
    const float* q_w1  = (const float*)d_in[19]; const float* q_b1  = (const float*)d_in[20];
    const float* q_w2  = (const float*)d_in[21]; const float* q_b2  = (const float*)d_in[22];
    float* out = (float*)d_out;

    // Workspace layout (~175 MB)
    float* ws = (float*)d_ws;
    size_t o = 0;
    float* hEnc   = ws + o; o += (size_t)NN * HHH;   // encoder output [N,32]
    float* qb     = ws + o; o += (size_t)NN * HDD;
    float* kb     = ws + o; o += (size_t)NN * HDD;
    float* vb     = ws + o; o += (size_t)NN * HDD;
    float* h2     = ws + o; o += (size_t)NN * HDD;   // conv1 output
    float* h3     = ws + o; o += (size_t)NN * HDD;   // conv2 output
    float* accum  = ws + o; o += (size_t)NN * HDD;
    float* logits = ws + o; o += (size_t)EE * 4;
    float* mbuf   = ws + o; o += (size_t)NN * 4;
    float* denom  = ws + o; o += (size_t)NN * 4;
    float* tmp    = accum;   // encoder intermediate reuses accum (free then)

    const dim3 w32(32, 1, 1);
    const int rowT = NN / 16;                        // 3125 exact

    // Encoder MLP: x[N,128] -> relu -> [N,32] -> relu -> hEnc[N,32]
    gemm_bf16_wmma<<<dim3(rowT, HHH / 16), w32, 0, stream>>>(x,   enc_w1, enc_b1, tmp,  F_INN, HHH, 1);
    gemm_bf16_wmma<<<dim3(rowT, HHH / 16), w32, 0, stream>>>(tmp, enc_w2, enc_b2, hEnc, HHH,   HHH, 1);

    const int edgeBlocks = (EE * 32) / 256;          // exact (E % 8 == 0)

    // ---- Conv1 (input hEnc, K=32) ----
    gemm_bf16_wmma<<<dim3(rowT, HDD / 16), w32, 0, stream>>>(hEnc, c1_wq, c1_bq, qb, HHH, HDD, 0);
    gemm_bf16_wmma<<<dim3(rowT, HDD / 16), w32, 0, stream>>>(hEnc, c1_wk, c1_bk, kb, HHH, HDD, 0);
    gemm_bf16_wmma<<<dim3(rowT, HDD / 16), w32, 0, stream>>>(hEnc, c1_wv, c1_bv, vb, HHH, HDD, 0);
    hipMemsetAsync(accum, 0, (size_t)NN * HDD * sizeof(float), stream);
    hipMemsetAsync(denom, 0, (size_t)NN * 4 * sizeof(float), stream);
    init_neg_inf_kernel<<<(NN * 4 + 255) / 256, 256, 0, stream>>>(mbuf, NN * 4);
    edge_logits_kernel<<<edgeBlocks, 256, 0, stream>>>(qb, kb, ei, logits, mbuf);
    edge_accum_kernel<<<edgeBlocks, 256, 0, stream>>>(vb, ei, logits, mbuf, denom, accum);
    finalize_kernel<<<(NN * HDD + 255) / 256, 256, 0, stream>>>(accum, denom, h2);

    // ---- Conv2 (input h2, K=128) ----
    gemm_bf16_wmma<<<dim3(rowT, HDD / 16), w32, 0, stream>>>(h2, c2_wq, c2_bq, qb, HDD, HDD, 0);
    gemm_bf16_wmma<<<dim3(rowT, HDD / 16), w32, 0, stream>>>(h2, c2_wk, c2_bk, kb, HDD, HDD, 0);
    gemm_bf16_wmma<<<dim3(rowT, HDD / 16), w32, 0, stream>>>(h2, c2_wv, c2_bv, vb, HDD, HDD, 0);
    hipMemsetAsync(accum, 0, (size_t)NN * HDD * sizeof(float), stream);
    hipMemsetAsync(denom, 0, (size_t)NN * 4 * sizeof(float), stream);
    init_neg_inf_kernel<<<(NN * 4 + 255) / 256, 256, 0, stream>>>(mbuf, NN * 4);
    edge_logits_kernel<<<edgeBlocks, 256, 0, stream>>>(qb, kb, ei, logits, mbuf);
    edge_accum_kernel<<<edgeBlocks, 256, 0, stream>>>(vb, ei, logits, mbuf, denom, accum);
    finalize_kernel<<<(NN * HDD + 255) / 256, 256, 0, stream>>>(accum, denom, h3);

    // ---- Q-head MLP ----
    qmlp_kernel<<<BBB, 128, 0, stream>>>(hEnc, h2, h3, idx, q_w1, q_b1, q_w2, q_b2, out);

    (void)in_sizes; (void)n_in; (void)out_size; (void)ws_size;
}
